// Bidirectional_3015067041912
// MI455X (gfx1250) — matrix operand order repaired and checked by reading, not compiled, not run on hardware
//
#include <hip/hip_runtime.h>
#include <hip/hip_bf16.h>
#include <math.h>

// Problem constants (match reference)
#define BB 32
#define SS 2048
#define DD 256
#define HH 256

typedef __attribute__((ext_vector_type(16))) __bf16 v16bf;
typedef __attribute__((ext_vector_type(8)))  __bf16 v8bf;
typedef __attribute__((ext_vector_type(8)))  float  v8f;

// Hardware tanh (V_TANH_F32, gfx1250 TRANS op) — single-instruction activation
// on the serial critical path of the recurrence.
__device__ __forceinline__ float hw_tanh(float x) {
#if __has_builtin(__builtin_amdgcn_tanhf)
    return __builtin_amdgcn_tanhf(x);
#else
    return tanhf(x);
#endif
}

// ---------------------------------------------------------------------------
// Prep 1: X [B,S,D] f32 -> Xbf [S,B,D] bf16 (time-major: 16KB chunk per step)
// ---------------------------------------------------------------------------
__global__ void __launch_bounds__(256) prep_x(const float* __restrict__ x,
                                              __bf16* __restrict__ xbf) {
    size_t idx = (size_t)blockIdx.x * 256 + threadIdx.x;   // over S*B*D
    int d = (int)(idx & (DD - 1));
    int b = (int)((idx >> 8) & (BB - 1));
    int s = (int)(idx >> 13);
    float v = x[((size_t)b * SS + s) * DD + d];
    xbf[idx] = (__bf16)v;
}

// ---------------------------------------------------------------------------
// Prep 2: pack W = [Wx; Wh] (512 x 256) transposed to N-major bf16:
//   wpack[dir][n][k], k<256 -> Wx[k][n], else Wh[k-256][n]
// ---------------------------------------------------------------------------
__global__ void __launch_bounds__(256) prep_w(const float* __restrict__ Wx_f,
                                              const float* __restrict__ Wh_f,
                                              const float* __restrict__ Wx_b,
                                              const float* __restrict__ Wh_b,
                                              __bf16* __restrict__ wpack) {
    int idx = blockIdx.x * 256 + threadIdx.x;   // over 2*256*512
    int dir = idx >> 17;
    int rem = idx & 131071;
    int n = rem >> 9;
    int k = rem & 511;
    const float* Wx = dir ? Wx_b : Wx_f;
    const float* Wh = dir ? Wh_b : Wh_f;
    float v = (k < 256) ? Wx[(size_t)k * HH + n] : Wh[(size_t)(k - 256) * HH + n];
    wpack[(size_t)dir * (256 * 512) + (size_t)n * 512 + k] = (__bf16)v;
}

// ---------------------------------------------------------------------------
// A-fragment loader: 16-bit A 16x32 layout (ISA 7.12.2):
//  lanes 0-15 hold K {0..7, 16..23}, lanes 16-31 hold K {8..15, 24..31}
// ---------------------------------------------------------------------------
__device__ __forceinline__ v16bf load_a_frag(const __bf16* p) {
    v8bf lo = *(const v8bf*)(p);
    v8bf hi = *(const v8bf*)(p + 16);
    return __builtin_shufflevector(lo, hi, 0, 1, 2, 3, 4, 5, 6, 7,
                                           8, 9, 10, 11, 12, 13, 14, 15);
}

#define WMMA_BF16(A, B, C) __builtin_amdgcn_wmma_f32_16x16x32_bf16( \
    false, (A), false, (B), (short)0, (C), false, false)

// ---------------------------------------------------------------------------
// One recurrence step. xcur/xnxt are compile-time-distinct buffer pointers
// (time loop unrolled by 2 at the call site), so all LDS addresses fold to
// immediate offsets. The k-loop is a 2-stage pipeline with two live staging
// buffers (aA*, aB*) and four independent accumulator chains.
// ---------------------------------------------------------------------------
__device__ __forceinline__ void rnn_step(
    int s, int sn, bool prefetch_ok,
    const __bf16* __restrict__ xcur, __bf16* __restrict__ xnxt,
    __bf16* __restrict__ hb,
    const __bf16* __restrict__ xbf, float* __restrict__ out,
    const v16bf* barr, float bias_v,
    int tid, int arow0, int arow1, int hi16, int n_g, int dir)
{
    // software prefetch next x tile into registers (overlaps WMMA below)
    v16bf px = {};
    if (prefetch_ok)
        px = *(const v16bf*)(xbf + (size_t)sn * (BB * DD) + tid * 16);

    // four independent accumulator chains (even/odd k-blocks per M-tile)
    v8f c0a = {}, c0b = {}, c1a = {}, c1b = {};

    // prologue: stage k-blocks 0 and 1
    v16bf aA0 = load_a_frag(xcur + arow0);
    v16bf aA1 = load_a_frag(xcur + arow1);
    v16bf aB0 = load_a_frag(xcur + 32 + arow0);
    v16bf aB1 = load_a_frag(xcur + 32 + arow1);

#pragma unroll
    for (int kb = 0; kb < 16; kb += 2) {
        c0a = WMMA_BF16(aA0, barr[kb], c0a);
        c1a = WMMA_BF16(aA1, barr[kb], c1a);
        if (kb + 2 < 16) {   // refill A-stage while B-stage computes
            const int kn = kb + 2;
            const __bf16* nb = (kn < 8) ? (xcur + kn * 32) : (hb + (kn - 8) * 32);
            aA0 = load_a_frag(nb + arow0);
            aA1 = load_a_frag(nb + arow1);
        }
        c0b = WMMA_BF16(aB0, barr[kb + 1], c0b);
        c1b = WMMA_BF16(aB1, barr[kb + 1], c1b);
        if (kb + 3 < 16) {   // refill B-stage while (next) A-stage computes
            const int kn = kb + 3;
            const __bf16* nb = (kn < 8) ? (xcur + kn * 32) : (hb + (kn - 8) * 32);
            aB0 = load_a_frag(nb + arow0);
            aB1 = load_a_frag(nb + arow1);
        }
    }

    __syncthreads();   // everyone finished reading hb for this step

    // h' = tanh(acc + b); write h' (bf16->LDS, WMMA C layout) and out (f32)
    const size_t outcol = (size_t)dir * HH + n_g;
#pragma unroll
    for (int r = 0; r < 8; ++r) {
        const int row0 = r + hi16 * 8;
        const float h0 = hw_tanh(c0a[r] + c0b[r] + bias_v);
        const float h1 = hw_tanh(c1a[r] + c1b[r] + bias_v);
        hb[row0 * DD + n_g]        = (__bf16)h0;
        hb[(row0 + 16) * DD + n_g] = (__bf16)h1;
        out[((size_t)row0 * SS + s) * (2 * HH) + outcol]        = h0;
        out[((size_t)(row0 + 16) * SS + s) * (2 * HH) + outcol] = h1;
    }

    // publish prefetched x tile into the other buffer
    *(v16bf*)(xnxt + tid * 16) = px;
    __syncthreads();
}

// ---------------------------------------------------------------------------
// grid = 2 (dir 0 = forward, dir 1 = backward), 512 threads (16 waves).
// Wave w owns output N-tile w; weights live in 128 VGPRs for the whole scan.
// ---------------------------------------------------------------------------
__global__ void __launch_bounds__(512) rnn_scan(const __bf16* __restrict__ xbf,
                                                const __bf16* __restrict__ wpack,
                                                const float* __restrict__ bias_f,
                                                const float* __restrict__ bias_b,
                                                float* __restrict__ out) {
    extern __shared__ __bf16 smem[];
    __bf16* xb0 = smem;                 // 32*256 bf16  (x tile, buffer 0)
    __bf16* xb1 = smem + 8192;          // 32*256 bf16  (x tile, buffer 1)
    __bf16* hb  = smem + 16384;         // 32*256 bf16  (recurrent h tile)

    const int tid  = threadIdx.x;
    const int lane = tid & 31;
    const int wv   = tid >> 5;          // 0..15 : N-tile id
    const int dir  = blockIdx.x;
    const bool fwd = (dir == 0);

    const int r0   = lane & 15;
    const int hi16 = lane >> 4;
    const int asw  = hi16 ? 8 : 0;      // A-fragment K swizzle offset
    const int bsw  = hi16 ? 8 : 0;      // B-fragment K offset
    const int n_g  = wv * 16 + r0;      // global output column 0..255

    const int arow0 = r0 * DD + asw;
    const int arow1 = (r0 + 16) * DD + asw;

    // ---- B fragments (weights) into registers: 16 x v16bf = 128 VGPRs ----
    const __bf16* wp = wpack + (size_t)dir * (256 * 512) + (size_t)n_g * 512;
    v16bf barr[16];
#pragma unroll
    for (int kb = 0; kb < 16; ++kb)
        barr[kb] = load_a_frag(wp + kb * 32 + bsw);

    const float bias_v = (fwd ? bias_f : bias_b)[n_g];

    // ---- init: h = 0, preload first x tile ----
    const int s_first = fwd ? 0 : (SS - 1);
    {
        v16bf z = {};
        *(v16bf*)(hb + tid * 16) = z;
        v16bf x0 = *(const v16bf*)(xbf + (size_t)s_first * (BB * DD) + tid * 16);
        *(v16bf*)(xb0 + tid * 16) = x0;
    }
    __syncthreads();

    // time loop unrolled by 2 -> xb0/xb1 are compile-time constants per call
    for (int tt = 0; tt < SS; tt += 2) {
        const int sA = fwd ? tt       : (SS - 1 - tt);
        const int sB = fwd ? (tt + 1) : (SS - 2 - tt);
        const int sC = fwd ? (tt + 2) : (SS - 3 - tt);

        rnn_step(sA, sB, true, xb0, xb1, hb, xbf, out,
                 barr, bias_v, tid, arow0, arow1, hi16, n_g, dir);
        rnn_step(sB, sC, (tt + 2) < SS, xb1, xb0, hb, xbf, out,
                 barr, bias_v, tid, arow0, arow1, hi16, n_g, dir);
    }
}

// ---------------------------------------------------------------------------
extern "C" void kernel_launch(void* const* d_in, const int* in_sizes, int n_in,
                              void* d_out, int out_size, void* d_ws, size_t ws_size,
                              hipStream_t stream) {
    const float* x    = (const float*)d_in[0];
    const float* Wx_f = (const float*)d_in[1];
    const float* Wh_f = (const float*)d_in[2];
    const float* b_f  = (const float*)d_in[3];
    const float* Wx_b = (const float*)d_in[4];
    const float* Wh_b = (const float*)d_in[5];
    const float* b_b  = (const float*)d_in[6];
    float* out = (float*)d_out;

    // workspace layout: [0, 512KB) packed bf16 weights; [1MB, 33MB) bf16 X
    __bf16* wpack = (__bf16*)d_ws;
    __bf16* xbf   = (__bf16*)((char*)d_ws + (1u << 20));

    prep_w<<<(2 * 256 * 512) / 256, 256, 0, stream>>>(Wx_f, Wh_f, Wx_b, Wh_b, wpack);
    prep_x<<<(size_t)BB * SS * DD / 256, 256, 0, stream>>>(x, xbf);

    const size_t lds_bytes = 3 * BB * DD * sizeof(__bf16);  // 48 KB
    rnn_scan<<<2, 512, lds_bytes, stream>>>(xbf, wpack, b_f, b_b, out);
}
